// TorchFeedForwardNetwork_58299886076360
// MI455X (gfx1250) — compile-verified, matrix operand order checked by reading, not run on hardware
//
#include <hip/hip_runtime.h>

typedef __attribute__((ext_vector_type(16))) _Float16 v16h;
typedef __attribute__((ext_vector_type(8)))  float    v8f;

#define B_BATCH 16384
#define N_IN    1024
#define LAYERS  8
#define WL      512
#define NNODES  (N_IN + LAYERS * WL)   // 5120

#define LDS_STRIDE 40                  // padded row stride in halves (80B = 20 banks)
#define SEG        (128 * LDS_STRIDE)  // one 128x32 tile (5120 halves)

// ---- gfx1250 async global->LDS path (bypasses VGPR staging) ----
#if defined(__has_builtin)
#  if __has_builtin(__builtin_amdgcn_global_load_async_to_lds_b128) && \
      __has_builtin(__builtin_amdgcn_s_wait_asynccnt)
#    define USE_ASYNC_LDS 1
#  endif
#endif
#ifndef USE_ASYNC_LDS
#  define USE_ASYNC_LDS 0
#endif

#if USE_ASYNC_LDS
typedef int v4i_vec __attribute__((vector_size(16)));
typedef __attribute__((address_space(1))) v4i_vec* glob_v4i;
typedef __attribute__((address_space(3))) v4i_vec* lds_v4i;
#endif

__device__ __forceinline__ void stage_b128(const _Float16* g, _Float16* l) {
#if USE_ASYNC_LDS
    __builtin_amdgcn_global_load_async_to_lds_b128(
        (glob_v4i)(void*)g, (lds_v4i)(void*)l, 0, 0);
#else
    *(uint4*)l = *(const uint4*)g;
#endif
}

__device__ __forceinline__ void stage_fence() {
#if USE_ASYNC_LDS
    __builtin_amdgcn_s_wait_asynccnt(0);
#endif
}

// ---------------- kernel 0: x(f32) -> values f16 cols [0,1024), 4 elems/thread ----------------
__global__ __launch_bounds__(256) void k_convert_x(const float* __restrict__ x,
                                                   _Float16* __restrict__ valuesH) {
    int i = (blockIdx.x * 256 + threadIdx.x) * 4;    // B*1024 elements
    int b = i >> 10, c = i & 1023;
    const float4 v = *(const float4*)(x + i);
    _Float16* dst = valuesH + (size_t)b * NNODES + c;
    dst[0] = (_Float16)v.x; dst[1] = (_Float16)v.y;
    dst[2] = (_Float16)v.z; dst[3] = (_Float16)v.w;
}

// ---------------- kernel 1: Wh = f16(W * mask), 4 elems/thread ----------------
__global__ __launch_bounds__(256) void k_prep_w(const float* __restrict__ W,
                                                const float* __restrict__ mask,
                                                _Float16* __restrict__ Wh) {
    int i = (blockIdx.x * 256 + threadIdx.x) * 4;    // L*512*512 elements
    const float4 w = *(const float4*)(W + i);
    const float4 m = *(const float4*)(mask + i);
    _Float16* dst = Wh + i;
    dst[0] = (_Float16)(w.x * m.x); dst[1] = (_Float16)(w.y * m.y);
    dst[2] = (_Float16)(w.z * m.z); dst[3] = (_Float16)(w.w * m.w);
}

// ---------------- per-layer gather: Ag[b][j] = valuesH[b][in_idx[j]] ----------------
__global__ __launch_bounds__(256) void k_gather(const _Float16* __restrict__ valuesH,
                                                const int* __restrict__ in_idx,
                                                _Float16* __restrict__ Ag) {
    int i = blockIdx.x * 256 + threadIdx.x;          // B*512 threads
    int b = i >> 9, j = i & 511;
    Ag[i] = valuesH[(size_t)b * NNODES + in_idx[j]];
}

// Build a 16x32 f16 WMMA fragment from an LDS tile stored [row][k], stride 40 halves.
// ISA 7.12.2: lanes 0-15 row=lane, K={0..7,16..23}; lanes 16-31 row=lane-16, K={8..15,24..31}
__device__ __forceinline__ v16h frag_from_lds(const _Float16* rowBase, int koff) {
    union { uint4 u[2]; v16h v; } r;
    r.u[0] = *(const uint4*)(rowBase + koff);
    r.u[1] = *(const uint4*)(rowBase + koff + 16);
    return r.v;
}

// ---------------- per-layer GEMM: h = relu(Ag @ Wh^T + bias) ----------------
// Block tile 128(M) x 128(N); 8 waves as 4(M) x 2(N), wave tile 32x64 (acc 2x4).
// Double-buffered LDS: stage chunk k+1 (async global->LDS) while computing chunk k.
__global__ __launch_bounds__(256) void k_layer_gemm(
        const _Float16* __restrict__ Ag,     // B x 512 row major (gathered inputs)
        const _Float16* __restrict__ Wh,     // 512(n) x 512(k) row major
        const float*    __restrict__ bias,   // 512
        _Float16* __restrict__ valuesH,      // B x 5120 f16 node values
        float*    __restrict__ out,          // B x 512 final output
        int outColBase, int isLast) {
    __shared__ __align__(16) _Float16 lds[2 * 2 * SEG];   // 40 KB: {A,B} x 2 buffers

    const int tid  = threadIdx.x;
    const int lane = tid & 31;
    const int wave = tid >> 5;
    const int wm   = wave & 3;               // 4 waves along M
    const int wn   = wave >> 2;              // 2 waves along N
    const int blockM = blockIdx.x * 128;
    const int blockN = blockIdx.y * 128;

    // staging: A and B tiles are both 128 rows x 32 halves; 2 threads/row
    const int sRow = tid >> 1;               // 0..127
    const int sK   = (tid & 1) * 16;         // 0 or 16 (halves)
    const _Float16* gA = Ag + (size_t)(blockM + sRow) * 512 + sK;
    const _Float16* gB = Wh + (size_t)(blockN + sRow) * 512 + sK;

    const int laneRow = lane & 15;
    const int laneK   = (lane >> 4) * 8;     // 0 or 8

    v8f acc[2][4] = {};

    // stage chunk 0 into buffer 0
    {
        _Float16* la = lds + sRow * LDS_STRIDE + sK;
        stage_b128(gA,     la);
        stage_b128(gA + 8, la + 8);
        stage_b128(gB,     la + SEG);
        stage_b128(gB + 8, la + SEG + 8);
    }
    stage_fence();
    __syncthreads();

    int buf = 0;
    for (int k0 = 0; k0 < 512; k0 += 32) {
        const int nbuf = buf ^ 1;
        // stage next chunk into the other buffer (overlaps with WMMAs below)
        if (k0 + 32 < 512) {
            const _Float16* nA = gA + k0 + 32;
            const _Float16* nB = gB + k0 + 32;
            _Float16* la = lds + nbuf * (2 * SEG) + sRow * LDS_STRIDE + sK;
            stage_b128(nA,     la);
            stage_b128(nA + 8, la + 8);
            stage_b128(nB,     la + SEG);
            stage_b128(nB + 8, la + SEG + 8);
            if (k0 + 64 < 512)
                __builtin_prefetch(nA + 32, 0, 2);     // global_prefetch 2 chunks ahead
        }

        // compute on current buffer
        const _Float16* A0 = lds + buf * (2 * SEG);
        const _Float16* B0 = A0 + SEG;

        v16h aF[2], bF[4];
#pragma unroll
        for (int i = 0; i < 2; ++i)
            aF[i] = frag_from_lds(A0 + (wm * 32 + i * 16 + laneRow) * LDS_STRIDE, laneK);
#pragma unroll
        for (int j = 0; j < 4; ++j)
            bF[j] = frag_from_lds(B0 + (wn * 64 + j * 16 + laneRow) * LDS_STRIDE, laneK);

#pragma unroll
        for (int i = 0; i < 2; ++i)
#pragma unroll
            for (int j = 0; j < 4; ++j)
                acc[i][j] = __builtin_amdgcn_wmma_f32_16x16x32_f16(
                                false, aF[i], false, bF[j],
                                (short)0, acc[i][j], false, false);

        stage_fence();     // async loads into next buffer complete
        __syncthreads();   // visible to all waves; current buffer free to overwrite
        buf = nbuf;
    }

    // epilogue: C/D layout = VGPR r -> M = r + 8*(lane>=16), N = lane&15
    const int mHalf = (lane >> 4) * 8;
    const int nCol  = lane & 15;
#pragma unroll
    for (int i = 0; i < 2; ++i) {
#pragma unroll
        for (int j = 0; j < 4; ++j) {
            const int colL = blockN + wn * 64 + j * 16 + nCol;   // 0..511
            const float bv = bias[colL];
#pragma unroll
            for (int r = 0; r < 8; ++r) {
                const int row = blockM + wm * 32 + i * 16 + mHalf + r;
                float v = fmaxf(acc[i][j][r] + bv, 0.0f);
                if (isLast)
                    out[(size_t)row * WL + colL] = v;
                else
                    valuesH[(size_t)row * NNODES + outColBase + colL] = (_Float16)v;
            }
        }
    }
}

extern "C" void kernel_launch(void* const* d_in, const int* in_sizes, int n_in,
                              void* d_out, int out_size, void* d_ws, size_t ws_size,
                              hipStream_t stream) {
    const float* x      = (const float*)d_in[0];   // B x 1024
    const float* W      = (const float*)d_in[1];   // L x 512 x 512
    const float* mask   = (const float*)d_in[2];   // L x 512 x 512
    const float* bias   = (const float*)d_in[3];   // L x 512
    const int*   in_idx = (const int*)d_in[4];     // L x 512
    float* out = (float*)d_out;                    // B x 512

    char* ws = (char*)d_ws;
    _Float16* valuesH = (_Float16*)ws;                                    // B*5120 f16
    size_t offs = (size_t)B_BATCH * NNODES * sizeof(_Float16);
    _Float16* Wh = (_Float16*)(ws + offs);                                // L*512*512 f16
    offs += (size_t)LAYERS * WL * WL * sizeof(_Float16);
    _Float16* Ag = (_Float16*)(ws + offs);                                // B*512 f16

    k_convert_x<<<(B_BATCH * N_IN) / (256 * 4), 256, 0, stream>>>(x, valuesH);
    k_prep_w<<<(LAYERS * WL * WL) / (256 * 4), 256, 0, stream>>>(W, mask, Wh);

    for (int l = 0; l < LAYERS; ++l) {
        k_gather<<<(B_BATCH * WL) / 256, 256, 0, stream>>>(valuesH, in_idx + l * WL, Ag);
        dim3 grid(B_BATCH / 128, WL / 128);
        k_layer_gemm<<<grid, 256, 0, stream>>>(Ag, Wh + (size_t)l * WL * WL,
                                               bias + l * WL, valuesH, out,
                                               N_IN + l * WL, (l == LAYERS - 1) ? 1 : 0);
    }
}
